// NTC_36060545417673
// MI455X (gfx1250) — matrix operand order repaired and checked by reading, not compiled, not run on hardware
//
#include <hip/hip_runtime.h>
#include <hip/hip_bf16.h>

typedef __attribute__((ext_vector_type(16))) _Float16 v16h;
typedef __attribute__((ext_vector_type(8)))  _Float16 v8h;
typedef __attribute__((ext_vector_type(8)))  float    v8f;
typedef __attribute__((ext_vector_type(4)))  unsigned int u32x4;
typedef __attribute__((ext_vector_type(8)))  int          i32x8;
typedef __attribute__((ext_vector_type(4)))  int          i32x4;

#define KCHUNK 128               // K elements staged per TDM transfer
#define LDS_BYTES (128 * KCHUNK * 2)

// ---------------------------------------------------------------------------
// Generic 5x5-conv / GEMM via WMMA f32_16x16x32_f16 with 2x2 register tiling
// and TDM-staged weight tiles in LDS.
//   X : activations, f16 channels-last rows gathered from [B, Hin, Win, Cin]
//   W : [ntaps, Cout, Cin] f16 (per tap, per output column, K-contiguous)
//   out: [Ptot, Cout] f16   (Ptot = B*Hout*Wout output positions)
// MODE: 0 = conv stride1, 1 = conv stride2,
//       2 = transposed conv stride2 (gather from 2x zero-dilated input),
//       3 = plain GEMM (ntaps must be 1)
// SQ  : square A elements after load (GDN divisive-norm einsum)
// Block: 256 threads = 8 waves as 2 row-subtiles x 4 col-subtiles.
// Per (tap, 128-K chunk): wave 0 issues TENSOR_LOAD_TO_LDS of the 128-col
// weight tile (s_wait_tensorcnt + barrier), then all waves read B fragments
// from LDS (ds_load_b128) while the A gather stays on the VMEM pipe.
// ---------------------------------------------------------------------------
template<int MODE, int SQ>
__global__ __launch_bounds__(256)
void conv5_wmma(const _Float16* __restrict__ X, const _Float16* __restrict__ W,
                const float* __restrict__ bias, _Float16* __restrict__ out,
                int Ptot, int Hin, int Win, int Cin,
                int Hout, int Wout, int Cout, int ntaps)
{
  extern __shared__ _Float16 Bs[];     // [128 cols][KCHUNK] staged weights

  const int wave   = threadIdx.x >> 5;
  const int rowSub = wave & 1;
  const int colSub = wave >> 1;
  const int lane   = threadIdx.x & 31;
  const int m      = lane & 15;        // A-frag row / B,C column within tile
  const int kg     = lane >> 4;        // K-group half of the wave
  const int p0     = blockIdx.x * 64 + rowSub * 32;
  const int HW     = Hout * Wout;
  const int colBase128 = blockIdx.y * 128;
  const int nrows  = (Cout - colBase128) < 128 ? (Cout - colBase128) : 128;

  // two A-fragment rows handled by this lane
  int pA[2]; bool rOK[2]; int bb_[2], oyA[2], oxA[2];
#pragma unroll
  for (int r = 0; r < 2; ++r) {
    int p = p0 + r * 16 + m;
    rOK[r] = p < Ptot;
    int pc = rOK[r] ? p : 0;
    pA[r] = pc;
    int bq = pc / HW, rem = pc - bq * HW;
    bb_[r] = bq; oyA[r] = rem / Wout; oxA[r] = rem - (rem / Wout) * Wout;
  }
  // two B/C columns handled by this lane
  const int nr0 = colSub * 32 + m;          // LDS-relative columns
  const int nr1 = colSub * 32 + 16 + m;
  const int n0  = colBase128 + nr0;         // global output channels
  const int n1  = colBase128 + nr1;

  v8f acc00 = {}, acc01 = {}, acc10 = {}, acc11 = {};

  for (int t = 0; t < ntaps; ++t) {
    const int ky = t / 5, kx = t - (t / 5) * 5;
    bool ok[2]; long base[2];
#pragma unroll
    for (int r = 0; r < 2; ++r) {
      bool o = rOK[r];
      long bs;
      if (MODE == 3) {
        bs = (long)pA[r] * Cin;
      } else if (MODE == 2) {
        // transposed conv stride 2: input zero-dilated by 2, pad (2,3)
        int ty = oyA[r] + ky - 2, tx = oxA[r] + kx - 2;
        o = o && ty >= 0 && tx >= 0 && !(ty & 1) && !(tx & 1)
              && (ty >> 1) < Hin && (tx >> 1) < Win;
        bs = (((long)bb_[r] * Hin + (ty >> 1)) * Win + (tx >> 1)) * (long)Cin;
      } else {
        const int s = (MODE == 1) ? 2 : 1;
        int iy = oyA[r] * s + ky - 2, ix = oxA[r] * s + kx - 2;
        o = o && iy >= 0 && iy < Hin && ix >= 0 && ix < Win;
        bs = (((long)bb_[r] * Hin + iy) * Win + ix) * (long)Cin;
      }
      ok[r] = o; base[r] = bs;
    }
    const _Float16* xr0 = X + base[0] + kg * 8;
    const _Float16* xr1 = X + base[1] + kg * 8;

    for (int kc = 0; kc < Cin; kc += KCHUNK) {
      int kw = Cin - kc; if (kw > KCHUNK) kw = KCHUNK;

      __syncthreads();   // previous chunk fully consumed before overwrite
#if __has_builtin(__builtin_amdgcn_tensor_load_to_lds)
      if (threadIdx.x < 32) {
        // Tensor DMA descriptor (ISA 8.3/8.4): 2-D tile, data_size = 2 bytes,
        // dim0 = kw contiguous K elems, dim1 = nrows rows of stride Cin.
        unsigned long ga =
            (unsigned long)(const void*)(W + ((long)t * Cout + colBase128) * Cin + kc);
        unsigned ldsa = (unsigned)(unsigned long long)(void*)Bs;
        u32x4 g0;
        g0[0] = 1u;                                   // count=1, user mode
        g0[1] = ldsa;                                 // lds_addr (bytes)
        g0[2] = (unsigned)(ga & 0xffffffffu);         // global_addr[31:0]
        g0[3] = (unsigned)((ga >> 32) & 0x01ffffffu)  // global_addr[56:32]
              | 0x80000000u;                          // type=2 ("image")
        i32x8 g1;
        g1[0] = 0x00010000;                           // data_size=1 -> 2B
        g1[1] = (int)((unsigned)kw << 16);            // tensor_dim0[15:0]
        g1[2] = (int)((unsigned)nrows << 16);         // tensor_dim1[15:0]
        g1[3] = (int)((unsigned)kw << 16);            // tile_dim0 = kw
        g1[4] = (int)(unsigned)nrows;                 // tile_dim1, tile_dim2=0
        g1[5] = (int)(unsigned)Cin;                   // tensor_dim0_stride
        g1[6] = 0; g1[7] = 0;                         // dim1_stride unused (2D)
        i32x4 gz4 = {0, 0, 0, 0};
        i32x8 gz8 = {0, 0, 0, 0, 0, 0, 0, 0};
        __builtin_amdgcn_tensor_load_to_lds(g0, g1, gz4, gz4, gz8, 0);
        __builtin_amdgcn_s_wait_tensorcnt(0);
      }
#else
      for (int idx = threadIdx.x; idx < nrows * kw; idx += 256) {
        int rr = idx / kw, cc = idx - rr * kw;
        Bs[idx] = W[((long)t * Cout + colBase128 + rr) * Cin + kc + cc];
      }
#endif
      __syncthreads();   // tile visible to all waves

      const _Float16* bp0 = Bs + nr0 * kw + kg * 8;
      const _Float16* bp1 = Bs + nr1 * kw + kg * 8;

      for (int k0 = 0; k0 < kw; k0 += 32) {
        v8h a0lo = {}, a0hi = {}, a1lo = {}, a1hi = {};
        if (ok[0]) { a0lo = *(const v8h*)(xr0 + kc + k0);
                     a0hi = *(const v8h*)(xr0 + kc + k0 + 16); }
        if (ok[1]) { a1lo = *(const v8h*)(xr1 + kc + k0);
                     a1hi = *(const v8h*)(xr1 + kc + k0 + 16); }
        if (SQ) {
#pragma unroll
          for (int i = 0; i < 8; ++i) {
            a0lo[i] *= a0lo[i]; a0hi[i] *= a0hi[i];
            a1lo[i] *= a1lo[i]; a1hi[i] *= a1hi[i];
          }
        }
        v8h b0lo = *(const v8h*)(bp0 + k0);        // ds_load_b128
        v8h b0hi = *(const v8h*)(bp0 + k0 + 16);
        v8h b1lo = *(const v8h*)(bp1 + k0);
        v8h b1hi = *(const v8h*)(bp1 + k0 + 16);
        v16h A0, A1, B0, B1;
#pragma unroll
        for (int i = 0; i < 8; ++i) {
          A0[i] = a0lo[i]; A0[i + 8] = a0hi[i];
          A1[i] = a1lo[i]; A1[i + 8] = a1hi[i];
          B0[i] = b0lo[i]; B0[i + 8] = b0hi[i];
          B1[i] = b1lo[i]; B1[i + 8] = b1hi[i];
        }
        acc00 = __builtin_amdgcn_wmma_f32_16x16x32_f16(false, A0, false, B0,
                                                       (short)0, acc00, false, false);
        acc01 = __builtin_amdgcn_wmma_f32_16x16x32_f16(false, A0, false, B1,
                                                       (short)0, acc01, false, false);
        acc10 = __builtin_amdgcn_wmma_f32_16x16x32_f16(false, A1, false, B0,
                                                       (short)0, acc10, false, false);
        acc11 = __builtin_amdgcn_wmma_f32_16x16x32_f16(false, A1, false, B1,
                                                       (short)0, acc11, false, false);
      }
    }
  }

  const int n0c = n0 < Cout ? n0 : Cout - 1;
  const int n1c = n1 < Cout ? n1 : Cout - 1;
  const float bia0 = bias ? bias[n0c] : 0.0f;
  const float bia1 = bias ? bias[n1c] : 0.0f;
#pragma unroll
  for (int v = 0; v < 8; ++v) {
    const int pr0 = p0 + kg * 8 + v;          // C layout: row = kg*8 + v
    const int pr1 = p0 + 16 + kg * 8 + v;
    if (pr0 < Ptot) {
      if (n0 < Cout) out[(long)pr0 * Cout + n0] = (_Float16)(acc00[v] + bia0);
      if (n1 < Cout) out[(long)pr0 * Cout + n1] = (_Float16)(acc01[v] + bia1);
    }
    if (pr1 < Ptot) {
      if (n0 < Cout) out[(long)pr1 * Cout + n0] = (_Float16)(acc10[v] + bia0);
      if (n1 < Cout) out[(long)pr1 * Cout + n1] = (_Float16)(acc11[v] + bia1);
    }
  }
}

// ---------------------------------------------------------------------------
// GDN apply: out = x * rsqrt(norm+beta)  (or * sqrt for inverse GDN)
// ---------------------------------------------------------------------------
__global__ void gdn_apply(const _Float16* __restrict__ X,
                          const _Float16* __restrict__ NRM,
                          const float* __restrict__ beta,
                          _Float16* __restrict__ out,
                          int C, long total, int inverse)
{
  long t = (long)blockIdx.x * blockDim.x + threadIdx.x;
  if (t >= total) return;
  int ch = (int)(t % C);
  float nrm = (float)NRM[t] + beta[ch];
  float x = (float)X[t];
  float r = inverse ? sqrtf(nrm) : rsqrtf(nrm);
  out[t] = (_Float16)(x * r);
}

// ---------------------------------------------------------------------------
// conv1: 2->256 stride2 (Cin too small for WMMA); x NCHW f32 -> f16 CL out
// ---------------------------------------------------------------------------
__global__ void conv1_kernel(const float* __restrict__ x, const float* __restrict__ w,
                             const float* __restrict__ bias, _Float16* __restrict__ out)
{
  int p = blockIdx.x;              // 0 .. 1024*16*16-1
  int o = threadIdx.x;             // 0 .. 255
  int b = p >> 8, rem = p & 255, oy = rem >> 4, ox = rem & 15;
  float acc = bias[o];
  for (int ky = 0; ky < 5; ++ky) {
    int iy = oy * 2 + ky - 2; if (iy < 0 || iy >= 32) continue;
    for (int kx = 0; kx < 5; ++kx) {
      int ix = ox * 2 + kx - 2; if (ix < 0 || ix >= 32) continue;
      for (int c = 0; c < 2; ++c)
        acc += x[((b * 2 + c) * 32 + iy) * 32 + ix] * w[((o * 2 + c) * 5 + ky) * 5 + kx];
    }
  }
  out[(long)p * 256 + o] = (_Float16)acc;
}

// ---------------------------------------------------------------------------
// Triangular stage: w_res = (I-L)y, + noise, likelihoods (erfc), then
// forward substitution y_hat = (I-L)^{-1} w_hat.  One thread per (channel,b);
// lanes share a channel so H loads broadcast (H is L2 resident, 5 MB).
// ---------------------------------------------------------------------------
__global__ void tri_kernel(const _Float16* __restrict__ Y, const float* __restrict__ noise,
                           const float* __restrict__ H, const float* __restrict__ mean,
                           const float* __restrict__ scale, float* __restrict__ lik,
                           _Float16* __restrict__ Yhat)
{
  const int C = 320, B = 1024;
  int t = blockIdx.x * blockDim.x + threadIdx.x;
  if (t >= B * C) return;
  int c = t >> 10;        // / 1024
  int b = t & 1023;
  float y[64];
  for (int i = 0; i < 64; ++i) y[i] = (float)Y[((long)b * 64 + i) * C + c];
  const float* Hc = H + (long)c * 4096;
  const float cneg = -0.70710678118654752f;   // -1/sqrt(2)
  // descending: w_res[i] = y[i] - sum_{j<i} L[i,j] y[j]  (y[j<i] still original)
  for (int i = 63; i >= 0; --i) {
    float s = y[i];
    for (int j = 0; j < i; ++j) s -= Hc[i * 64 + j] * y[j];
    float wh = s + noise[((long)b * C + c) * 64 + i];
    float sc = scale[c * 64 + i]; sc = sc < 0.11f ? 0.11f : sc;
    float v  = fabsf(wh - mean[c * 64 + i]);
    float up = 0.5f * erfcf(cneg * (( 0.5f - v) / sc));
    float lo = 0.5f * erfcf(cneg * ((-0.5f - v) / sc));
    float l = up - lo; l = l < 1e-9f ? 1e-9f : l;
    lik[((long)b * C + c) * 64 + i] = l;
    y[i] = wh;
  }
  // unit-lower forward substitution: y_hat[i] = w_hat[i] + sum_{j<i} L[i,j] y_hat[j]
  for (int i = 0; i < 64; ++i) {
    float s = y[i];
    for (int j = 0; j < i; ++j) s += Hc[i * 64 + j] * y[j];
    y[i] = s;
    Yhat[((long)b * 64 + i) * C + c] = (_Float16)s;
  }
}

// ---------------------------------------------------------------------------
// deconv3: 256->2, transposed stride2 (Cout too small for WMMA), clip, NCHW f32
// ---------------------------------------------------------------------------
__global__ void deconv3_kernel(const _Float16* __restrict__ D, const float* __restrict__ w,
                               const float* __restrict__ bias, float* __restrict__ xhat)
{
  int t = blockIdx.x * blockDim.x + threadIdx.x;   // over 1024*32*32*2
  if (t >= 1024 * 32 * 32 * 2) return;
  int o  = t & 1;
  int ox = (t >> 1) & 31;
  int oy = (t >> 6) & 31;
  int b  = t >> 11;
  float acc = bias[o];
  for (int ky = 0; ky < 5; ++ky) {
    int ty = oy + ky - 2;
    if (ty < 0 || (ty & 1) || (ty >> 1) >= 16) continue;
    int iy = ty >> 1;
    for (int kx = 0; kx < 5; ++kx) {
      int tx = ox + kx - 2;
      if (tx < 0 || (tx & 1) || (tx >> 1) >= 16) continue;
      int ix = tx >> 1;
      const _Float16* row = D + ((long)b * 256 + iy * 16 + ix) * 256;
      int widx = o * 25 + (4 - ky) * 5 + (4 - kx);   // dw3[c][o][4-ky][4-kx]
      for (int c = 0; c < 256; ++c)
        acc += (float)row[c] * w[c * 50 + widx];
    }
  }
  acc = acc < 0.f ? 0.f : (acc > 1.f ? 1.f : acc);
  xhat[((long)b * 2 + o) * 1024 + (oy * 32 + ox)] = acc;
}

// ---------------------------------------------------------------------------
// Weight prep kernels: f32 OIHW / IOHW -> f16 [tap][Cout][Cin] (flip for deconv)
// ---------------------------------------------------------------------------
__global__ void prep_fwd_w(const float* __restrict__ w, _Float16* __restrict__ wt,
                           int Cout, int Cin)
{
  long t = (long)blockIdx.x * blockDim.x + threadIdx.x;
  long total = 25L * Cout * Cin;
  if (t >= total) return;
  int c   = (int)(t % Cin);
  int o   = (int)((t / Cin) % Cout);
  int tap = (int)(t / ((long)Cin * Cout));
  int ky = tap / 5, kx = tap % 5;
  wt[t] = (_Float16)w[((o * Cin + c) * 5 + ky) * 5 + kx];
}

__global__ void prep_dec_w(const float* __restrict__ w, _Float16* __restrict__ wt,
                           int Cin, int Cout)
{
  long t = (long)blockIdx.x * blockDim.x + threadIdx.x;
  long total = 25L * Cout * Cin;
  if (t >= total) return;
  int c   = (int)(t % Cin);
  int o   = (int)((t / Cin) % Cout);
  int tap = (int)(t / ((long)Cin * Cout));
  int ky = tap / 5, kx = tap % 5;
  wt[t] = (_Float16)w[((c * Cout + o) * 5 + (4 - ky)) * 5 + (4 - kx)];
}

__global__ void f32_to_f16_k(const float* __restrict__ a, _Float16* __restrict__ o, long n)
{
  long t = (long)blockIdx.x * blockDim.x + threadIdx.x;
  if (t < n) o[t] = (_Float16)a[t];
}

// ---------------------------------------------------------------------------
extern "C" void kernel_launch(void* const* d_in, const int* in_sizes, int n_in,
                              void* d_out, int out_size, void* d_ws, size_t ws_size,
                              hipStream_t stream)
{
  (void)in_sizes; (void)n_in; (void)out_size; (void)ws_size;
  const float* x       = (const float*)d_in[0];
  const float* noise   = (const float*)d_in[1];
  const float* w1      = (const float*)d_in[2];
  const float* b1      = (const float*)d_in[3];
  const float* beta1   = (const float*)d_in[4];
  const float* gamma1  = (const float*)d_in[5];
  const float* w2      = (const float*)d_in[6];
  const float* b2      = (const float*)d_in[7];
  const float* beta2   = (const float*)d_in[8];
  const float* gamma2  = (const float*)d_in[9];
  const float* w3      = (const float*)d_in[10];
  const float* b3      = (const float*)d_in[11];
  const float* dw1     = (const float*)d_in[12];
  const float* db1     = (const float*)d_in[13];
  const float* ibeta1  = (const float*)d_in[14];
  const float* igamma1 = (const float*)d_in[15];
  const float* dw2     = (const float*)d_in[16];
  const float* db2     = (const float*)d_in[17];
  const float* ibeta2  = (const float*)d_in[18];
  const float* igamma2 = (const float*)d_in[19];
  const float* dw3     = (const float*)d_in[20];
  const float* db3     = (const float*)d_in[21];
  const float* meanp   = (const float*)d_in[22];
  const float* scalep  = (const float*)d_in[23];
  const float* Hmat    = (const float*)d_in[24];

  char* ws = (char*)d_ws;
  size_t off = 0;
  auto alloc = [&](size_t bytes) -> char* {
    char* p = ws + off; off += (bytes + 255) & ~(size_t)255; return p;
  };
  const long P1 = 1024L * 16 * 16;   // 262144 positions (16x16 maps)
  const long P2 = 1024L * 8 * 8;     //  65536 positions (8x8 maps)

  _Float16* bufA = (_Float16*)alloc((size_t)P1 * 256 * 2); // conv1 out / deconv2 out
  _Float16* bufB = (_Float16*)alloc((size_t)P1 * 256 * 2); // norm scratch (all GDNs)
  _Float16* bufC = (_Float16*)alloc((size_t)P1 * 256 * 2); // gdn1 out / igdn2 out
  _Float16* bufD = (_Float16*)alloc((size_t)P2 * 256 * 2); // conv2 out / deconv1 out
  _Float16* bufE = (_Float16*)alloc((size_t)P2 * 256 * 2); // gdn2 out / igdn1 out
  _Float16* bufF = (_Float16*)alloc((size_t)P2 * 320 * 2); // conv3 out (y)
  _Float16* bufG = (_Float16*)alloc((size_t)P2 * 320 * 2); // y_hat
  _Float16* WT2  = (_Float16*)alloc(25UL * 256 * 256 * 2);
  _Float16* WT3  = (_Float16*)alloc(25UL * 320 * 256 * 2);
  _Float16* WD1  = (_Float16*)alloc(25UL * 256 * 320 * 2);
  _Float16* WD2  = (_Float16*)alloc(25UL * 256 * 256 * 2);
  _Float16* G1h  = (_Float16*)alloc(256UL * 256 * 2);
  _Float16* G2h  = (_Float16*)alloc(256UL * 256 * 2);
  _Float16* IG1h = (_Float16*)alloc(256UL * 256 * 2);
  _Float16* IG2h = (_Float16*)alloc(256UL * 256 * 2);

  // ---- weight prep (f32 -> f16, tap-major, K-contiguous per output column)
  {
    long n2 = 25L * 256 * 256, n3 = 25L * 320 * 256, ng = 256L * 256;
    prep_fwd_w<<<dim3((unsigned)((n2 + 255) / 256)), 256, 0, stream>>>(w2, WT2, 256, 256);
    prep_fwd_w<<<dim3((unsigned)((n3 + 255) / 256)), 256, 0, stream>>>(w3, WT3, 320, 256);
    prep_dec_w<<<dim3((unsigned)((n3 + 255) / 256)), 256, 0, stream>>>(dw1, WD1, 320, 256);
    prep_dec_w<<<dim3((unsigned)((n2 + 255) / 256)), 256, 0, stream>>>(dw2, WD2, 256, 256);
    f32_to_f16_k<<<dim3((unsigned)((ng + 255) / 256)), 256, 0, stream>>>(gamma1,  G1h,  ng);
    f32_to_f16_k<<<dim3((unsigned)((ng + 255) / 256)), 256, 0, stream>>>(gamma2,  G2h,  ng);
    f32_to_f16_k<<<dim3((unsigned)((ng + 255) / 256)), 256, 0, stream>>>(igamma1, IG1h, ng);
    f32_to_f16_k<<<dim3((unsigned)((ng + 255) / 256)), 256, 0, stream>>>(igamma2, IG2h, ng);
  }

  const unsigned gx1 = (unsigned)((P1 + 63) / 64);   // 4096 row-tiles of 64
  const unsigned gx2 = (unsigned)((P2 + 63) / 64);   // 1024
  const unsigned gy256 = 2;                          // ceil(256/128)
  const unsigned gy320 = 3;                          // ceil(320/128), tail clamped
  const size_t   shb   = LDS_BYTES;                  // 32 KB staged weight tile

  // ---- analysis transform
  conv1_kernel<<<dim3(262144), 256, 0, stream>>>(x, w1, b1, bufA);
  conv5_wmma<3, 1><<<dim3(gx1, gy256), 256, shb, stream>>>(bufA, G1h, nullptr, bufB,
      (int)P1, 16, 16, 256, 16, 16, 256, 1);
  gdn_apply<<<dim3((unsigned)((P1 * 256 + 255) / 256)), 256, 0, stream>>>(
      bufA, bufB, beta1, bufC, 256, P1 * 256, 0);

  conv5_wmma<1, 0><<<dim3(gx2, gy256), 256, shb, stream>>>(bufC, WT2, b2, bufD,
      (int)P2, 16, 16, 256, 8, 8, 256, 25);
  conv5_wmma<3, 1><<<dim3(gx2, gy256), 256, shb, stream>>>(bufD, G2h, nullptr, bufB,
      (int)P2, 8, 8, 256, 8, 8, 256, 1);
  gdn_apply<<<dim3((unsigned)((P2 * 256 + 255) / 256)), 256, 0, stream>>>(
      bufD, bufB, beta2, bufE, 256, P2 * 256, 0);

  conv5_wmma<0, 0><<<dim3(gx2, gy320), 256, shb, stream>>>(bufE, WT3, b3, bufF,
      (int)P2, 8, 8, 256, 8, 8, 320, 25);

  // ---- triangular transform + likelihoods (lik goes straight to d_out)
  float* lik = (float*)d_out + 1024L * 2 * 32 * 32;
  tri_kernel<<<dim3(1280), 256, 0, stream>>>(bufF, noise, Hmat, meanp, scalep, lik, bufG);

  // ---- synthesis transform
  conv5_wmma<0, 0><<<dim3(gx2, gy256), 256, shb, stream>>>(bufG, WD1, db1, bufD,
      (int)P2, 8, 8, 320, 8, 8, 256, 25);
  conv5_wmma<3, 1><<<dim3(gx2, gy256), 256, shb, stream>>>(bufD, IG1h, nullptr, bufB,
      (int)P2, 8, 8, 256, 8, 8, 256, 1);
  gdn_apply<<<dim3((unsigned)((P2 * 256 + 255) / 256)), 256, 0, stream>>>(
      bufD, bufB, ibeta1, bufE, 256, P2 * 256, 1);

  conv5_wmma<2, 0><<<dim3(gx1, gy256), 256, shb, stream>>>(bufE, WD2, db2, bufA,
      (int)P1, 8, 8, 256, 16, 16, 256, 25);
  conv5_wmma<3, 1><<<dim3(gx1, gy256), 256, shb, stream>>>(bufA, IG2h, nullptr, bufB,
      (int)P1, 16, 16, 256, 16, 16, 256, 1);
  gdn_apply<<<dim3((unsigned)((P1 * 256 + 255) / 256)), 256, 0, stream>>>(
      bufA, bufB, ibeta2, bufC, 256, P1 * 256, 1);

  deconv3_kernel<<<dim3((unsigned)((1024L * 32 * 32 * 2 + 255) / 256)), 256, 0, stream>>>(
      bufC, dw3, db3, (float*)d_out);
}
